// SSFE_Net_670014898798
// MI455X (gfx1250) — matrix operand order repaired and checked by reading, not compiled, not run on hardware
//
#include <hip/hip_runtime.h>
#include <hip/hip_bf16.h>
#include <stdint.h>

typedef __attribute__((ext_vector_type(16))) _Float16 v16h;
typedef __attribute__((ext_vector_type(8)))  _Float16 v8h;
typedef __attribute__((ext_vector_type(8)))  float    v8f;

// ---------------------------------------------------------------------------
// WMMA helpers (CDNA5 gfx1250, wave32).
// A (16x32 f16): lane = M (lanes 0-15 / 16-31 mirror), halves e -> K=(lane>>4)*16+e
//                => two contiguous 8-half (16B) loads.
// B (32x16 f16): lane&15 = N column, halves e -> K=(lane>>4)*16+e => one contiguous 16-half (32B) load.
// C/D (16x16 f32): VGPR r: M = r + (lane>>4)*8, N = lane&15.
// ---------------------------------------------------------------------------

__device__ __forceinline__ v8f wmma16x16x32(v16h a, v16h b, v8f c) {
  return __builtin_amdgcn_wmma_f32_16x16x32_f16(false, a, false, b, (short)0, c,
                                                false, false);
}

__device__ __forceinline__ v16h load_frag_A(const _Float16* base, int rowStride, int lane) {
  const _Float16* pr = base + (size_t)(lane & 15) * rowStride + ((lane >> 4) << 4);
  union { v16h v; v8h h[2]; } u;
  u.h[0] = *(const v8h*)(pr);
  u.h[1] = *(const v8h*)(pr + 8);
  return u.v;
}

__device__ __forceinline__ v16h load_frag_B(const _Float16* w, int Cin, int ntile, int kc, int lane) {
  const _Float16* pr = w + (size_t)(ntile * 16 + (lane & 15)) * Cin + kc * 32 + ((lane >> 4) << 4);
  return *(const v16h*)(pr);
}

template<bool LEAKY>
__device__ __forceinline__ float actf(float v) {
  if (LEAKY) return v > 0.f ? v : 0.2f * v;
  return v > 0.f ? v : 0.f;
}

// ---------------------------------------------------------------------------
// Prep: weight f32->f16, BN -> (scale, shift)
// ---------------------------------------------------------------------------
__global__ void cvt_f16_kernel(const float* __restrict__ in, _Float16* __restrict__ out, int n) {
  int i = blockIdx.x * blockDim.x + threadIdx.x;
  if (i < n) out[i] = (_Float16)in[i];
}

__global__ void bn_prep_kernel(const float* __restrict__ p, float* __restrict__ scsh, int C) {
  int i = blockIdx.x * blockDim.x + threadIdx.x;
  if (i < C) {
    float g = p[i], b = p[C + i], m = p[2 * C + i], v = p[3 * C + i];
    float s = g * rsqrtf(v + 1e-5f);
    scsh[i] = s;
    scsh[C + i] = b - m * s;
  }
}

// ---------------------------------------------------------------------------
// f_o = relu(bn1(w1 @ x)), stored point-major (B, N, 64)
// ---------------------------------------------------------------------------
__global__ void pointwise1_kernel(const float* __restrict__ x, const float* __restrict__ w1,
                                  const float* __restrict__ scsh, float* __restrict__ f_t,
                                  int B, int N) {
  int gid = blockIdx.x * blockDim.x + threadIdx.x;
  if (gid >= B * N) return;
  int b = gid / N, n = gid % N;
  float xi[6];
#pragma unroll
  for (int c = 0; c < 6; ++c) xi[c] = x[((size_t)b * 6 + c) * N + n];
  float* orow = f_t + (size_t)gid * 64;
  for (int o = 0; o < 64; ++o) {
    float a = 0.f;
#pragma unroll
    for (int c = 0; c < 6; ++c) a += xi[c] * w1[o * 6 + c];
    float v = scsh[o] * a + scsh[64 + o];
    orow[o] = v > 0.f ? v : 0.f;
  }
}

__global__ void extract_xyz_kernel(const float* __restrict__ x, float* __restrict__ xyz,
                                   int B, int N) {
  int gid = blockIdx.x * blockDim.x + threadIdx.x;
  if (gid >= B * N) return;
  int b = gid / N, n = gid % N;
#pragma unroll
  for (int d = 0; d < 3; ++d) xyz[(size_t)gid * 3 + d] = x[((size_t)b * 6 + d) * N + n];
}

// ---------------------------------------------------------------------------
// KNN: one thread per query, register insertion-sorted top-K (smallest dist).
// ---------------------------------------------------------------------------
template<int K, int D>
__global__ void knn_kernel(const float* __restrict__ query, const float* __restrict__ ref,
                           int Q, int R, int* __restrict__ outIdx, int total) {
  int gid = blockIdx.x * blockDim.x + threadIdx.x;
  if (gid >= total) return;
  int b = gid / Q;
  const float* qrow = query + (size_t)gid * D;
  float qv[D];
#pragma unroll
  for (int c = 0; c < D; ++c) qv[c] = qrow[c];
  float bd[K]; int bi[K];
#pragma unroll
  for (int t = 0; t < K; ++t) { bd[t] = 3.0e38f; bi[t] = 0; }
  const float* rb = ref + (size_t)b * R * D;
  for (int r = 0; r < R; ++r) {
    const float* rr = rb + (size_t)r * D;
    float d = 0.f;
#pragma unroll
    for (int c = 0; c < D; ++c) { float t = qv[c] - rr[c]; d += t * t; }
    if (d < bd[K - 1]) {
      bd[K - 1] = d; bi[K - 1] = r;
#pragma unroll
      for (int t = K - 1; t > 0; --t) {
        if (bd[t] < bd[t - 1]) {
          float td = bd[t]; bd[t] = bd[t - 1]; bd[t - 1] = td;
          int ti = bi[t]; bi[t] = bi[t - 1]; bi[t - 1] = ti;
        }
      }
    }
  }
  int* orow = outIdx + (size_t)gid * K;
#pragma unroll
  for (int t = 0; t < K; ++t) orow[t] = bi[t];
}

// ---------------------------------------------------------------------------
// Farthest point sampling: one block per batch.
// ---------------------------------------------------------------------------
__global__ void fps_kernel(const float* __restrict__ pts, int N, int npoint,
                           int* __restrict__ outIdx) {
  __shared__ float dists[2048];
  __shared__ float smax[256];
  __shared__ int   simax[256];
  __shared__ int   sfar;
  int b = blockIdx.x, t = threadIdx.x;
  const float* pb = pts + (size_t)b * N * 3;
  for (int i = t; i < N; i += 256) dists[i] = 1e10f;
  if (t == 0) sfar = 0;
  __syncthreads();
  for (int it = 0; it < npoint; ++it) {
    int far = sfar;
    if (t == 0) outIdx[(size_t)b * npoint + it] = far;
    float cx = pb[far * 3 + 0], cy = pb[far * 3 + 1], cz = pb[far * 3 + 2];
    float lm = -1.f; int li = 0;
    for (int i = t; i < N; i += 256) {
      float dx = pb[i * 3 + 0] - cx, dy = pb[i * 3 + 1] - cy, dz = pb[i * 3 + 2] - cz;
      float d = dx * dx + dy * dy + dz * dz;
      float nd = fminf(dists[i], d);
      dists[i] = nd;
      if (nd > lm) { lm = nd; li = i; }
    }
    smax[t] = lm; simax[t] = li;
    __syncthreads();
    for (int s = 128; s > 0; s >>= 1) {
      if (t < s && smax[t + s] > smax[t]) { smax[t] = smax[t + s]; simax[t] = simax[t + s]; }
      __syncthreads();
    }
    if (t == 0) sfar = simax[0];
    __syncthreads();
  }
}

__global__ void gather_rows_kernel(const float* __restrict__ src, const int* __restrict__ idx,
                                   float* __restrict__ dst, int Nsrc, int Q, int D, int total) {
  int gid = blockIdx.x * blockDim.x + threadIdx.x;
  if (gid >= total) return;
  int b = gid / Q;
  int r = idx[gid];
  const float* s = src + ((size_t)b * Nsrc + r) * D;
  float* d = dst + (size_t)gid * D;
  for (int c = 0; c < D; ++c) d[c] = s[c];
}

__global__ void skip_gather_kernel(const float* __restrict__ f_t, _Float16* __restrict__ skip_h,
                                   int total) {
  int gid = blockIdx.x * blockDim.x + threadIdx.x;
  if (gid >= total) return;
  int c = gid & 63;
  int s = (gid >> 6) & 255;
  int b = gid >> 14;
  skip_h[gid] = (_Float16)f_t[(((size_t)b * 2048) + (size_t)s * 8) * 64 + c];
}

// ---------------------------------------------------------------------------
// rows = [feat[idx[j]] - center, center] (32 rows, padded), WMMA x W^T,
// epilogue: act(bn(.)) then max over valid rows. Covers edge_branch + local_op.
// Block = 4 waves, one output point per wave.
// ---------------------------------------------------------------------------
template<int C2, int O, int KR, bool LEAKY>
__global__ void edge_max_gemm(const float* __restrict__ feat, int Nsrc, int Nout,
                              const int* __restrict__ idx, const int* __restrict__ cidx,
                              const _Float16* __restrict__ wh, const float* __restrict__ scsh,
                              float* __restrict__ outf,
                              _Float16* __restrict__ outh, int hstride, int hoff) {
  constexpr int C = C2 / 2;
  __shared__ __align__(32) _Float16 eL[4][32][C2];
  int wid = threadIdx.x >> 5;
  int lane = threadIdx.x & 31;
  int b = blockIdx.y;
  int s = blockIdx.x * 4 + wid;

  const float* fb = feat + (size_t)b * Nsrc * C;
  int ci = cidx ? cidx[(size_t)b * Nout + s] : s;
  float cenv[C / 32];
#pragma unroll
  for (int k = 0; k < C / 32; ++k) cenv[k] = fb[(size_t)ci * C + lane + 32 * k];

  const int* irow = idx + ((size_t)b * Nout + s) * KR;
  for (int j = 0; j < 32; ++j) {
    if (j < KR) {
      int nb = irow[j];
#pragma unroll
      for (int k = 0; k < C / 32; ++k) {
        float nv = fb[(size_t)nb * C + lane + 32 * k];
        eL[wid][j][lane + 32 * k]     = (_Float16)(nv - cenv[k]);
        eL[wid][j][C + lane + 32 * k] = (_Float16)cenv[k];
      }
    } else {
#pragma unroll
      for (int k = 0; k < C / 32; ++k) {
        eL[wid][j][lane + 32 * k]     = (_Float16)0.f;
        eL[wid][j][C + lane + 32 * k] = (_Float16)0.f;
      }
    }
  }
  __syncthreads();

  for (int nt = 0; nt < O / 16; ++nt) {
    v8f acc0 = {}; v8f acc1 = {};
    for (int kc = 0; kc < C2 / 32; ++kc) {
      v16h bB = load_frag_B(wh, C2, nt, kc, lane);
      v16h a0 = load_frag_A(&eL[wid][0][kc * 32], C2, lane);
      v16h a1 = load_frag_A(&eL[wid][16][kc * 32], C2, lane);
      acc0 = wmma16x16x32(a0, bB, acc0);
      acc1 = wmma16x16x32(a1, bB, acc1);
    }
    int oo = nt * 16 + (lane & 15);
    float sc = scsh[oo], sh = scsh[O + oo];
    float mx = -3.0e38f;
#pragma unroll
    for (int r = 0; r < 8; ++r) {
      int m0 = r + ((lane >> 4) << 3);
      float v0 = actf<LEAKY>(sc * acc0[r] + sh);
      if (m0 < KR) mx = fmaxf(mx, v0);
      float v1 = actf<LEAKY>(sc * acc1[r] + sh);
      if (16 + m0 < KR) mx = fmaxf(mx, v1);
    }
    mx = fmaxf(mx, __shfl_xor(mx, 16, 32));
    if (lane < 16) {
      if (outf) outf[((size_t)b * Nout + s) * O + oo] = mx;
      if (outh) outh[((size_t)b * Nout + s) * hstride + hoff + oo] = (_Float16)mx;
    }
  }
}

// ---------------------------------------------------------------------------
// Plain point-major GEMM + BN + activation. One wave per 16-point M tile.
// Outputs optionally f32 point-major, f16 strided slice, f32 channel-major.
// ---------------------------------------------------------------------------
template<int Cin, int O, bool LEAKY>
__global__ void gemm_bn_act(const _Float16* __restrict__ in, int Nout,
                            const _Float16* __restrict__ wh, const float* __restrict__ scsh,
                            float* __restrict__ out_t,
                            _Float16* __restrict__ outh, int hstride, int hoff,
                            float* __restrict__ out_cm) {
  int b = blockIdx.y;
  int mbase = blockIdx.x * 16;
  int lane = threadIdx.x & 31;
  const _Float16* Ab = in + ((size_t)b * Nout + mbase) * Cin;
  for (int nt = 0; nt < O / 16; ++nt) {
    v8f acc = {};
    for (int kc = 0; kc < Cin / 32; ++kc) {
      v16h a = load_frag_A(Ab + kc * 32, Cin, lane);
      v16h bB = load_frag_B(wh, Cin, nt, kc, lane);
      acc = wmma16x16x32(a, bB, acc);
    }
    int oo = nt * 16 + (lane & 15);
    float sc = scsh[oo], sh = scsh[O + oo];
#pragma unroll
    for (int r = 0; r < 8; ++r) {
      int m = mbase + r + ((lane >> 4) << 3);
      float v = actf<LEAKY>(sc * acc[r] + sh);
      if (out_t)  out_t[((size_t)b * Nout + m) * O + oo] = v;
      if (outh)   outh[((size_t)b * Nout + m) * hstride + hoff + oo] = (_Float16)v;
      if (out_cm) out_cm[((size_t)b * O + oo) * Nout + m] = v;
    }
  }
}

// ---------------------------------------------------------------------------
// Host launcher
// ---------------------------------------------------------------------------
extern "C" void kernel_launch(void* const* d_in, const int* in_sizes, int n_in,
                              void* d_out, int out_size, void* d_ws, size_t ws_size,
                              hipStream_t stream) {
  (void)in_sizes; (void)n_in; (void)out_size; (void)ws_size;
  const int B = 16, N = 2048;
  const float* x       = (const float*)d_in[0];
  const float* w1      = (const float*)d_in[1];
  const float* bn1     = (const float*)d_in[2];
  const float* w2      = (const float*)d_in[3];
  const float* bn2     = (const float*)d_in[4];
  const float* geo_w   = (const float*)d_in[5];
  const float* geo_bn  = (const float*)d_in[6];
  const float* feat_w  = (const float*)d_in[7];
  const float* feat_bn = (const float*)d_in[8];
  const float* fuse_w  = (const float*)d_in[9];
  const float* fuse_bn = (const float*)d_in[10];
  const float* gl0_w   = (const float*)d_in[11];
  const float* gl0_bn  = (const float*)d_in[12];
  const float* gl1_w   = (const float*)d_in[13];
  const float* gl1_bn  = (const float*)d_in[14];
  const float* wf      = (const float*)d_in[15];
  const float* bnf     = (const float*)d_in[16];
  float* out = (float*)d_out;

  uint8_t* base = (uint8_t*)d_ws;
  size_t off = 0;
  auto alloc = [&](size_t bytes) -> void* {
    void* r = base + off;
    off += (bytes + 255) & ~(size_t)255;
    return r;
  };

  _Float16* wh_geo  = (_Float16*)alloc((size_t)64 * 128 * 2);
  _Float16* wh_feat = (_Float16*)alloc((size_t)64 * 128 * 2);
  _Float16* wh_fuse = (_Float16*)alloc((size_t)64 * 128 * 2);
  _Float16* wh_gl0  = (_Float16*)alloc((size_t)128 * 128 * 2);
  _Float16* wh_gl1  = (_Float16*)alloc((size_t)256 * 256 * 2);
  _Float16* wh_w2   = (_Float16*)alloc((size_t)1024 * 64 * 2);
  _Float16* wh_wf   = (_Float16*)alloc((size_t)512 * 1280 * 2);
  float* sc_bn1  = (float*)alloc(2 * 64 * 4);
  float* sc_bn2  = (float*)alloc(2 * 1024 * 4);
  float* sc_geo  = (float*)alloc(2 * 64 * 4);
  float* sc_feat = (float*)alloc(2 * 64 * 4);
  float* sc_fuse = (float*)alloc(2 * 64 * 4);
  float* sc_gl0  = (float*)alloc(2 * 128 * 4);
  float* sc_gl1  = (float*)alloc(2 * 256 * 4);
  float* sc_bnf  = (float*)alloc(2 * 512 * 4);
  float*    f_t       = (float*)alloc((size_t)B * N * 64 * 4);
  float*    xyz_t     = (float*)alloc((size_t)B * N * 3 * 4);
  int*      idx_g     = (int*)alloc((size_t)B * N * 20 * 4);
  int*      idx_f     = (int*)alloc((size_t)B * N * 20 * 4);
  _Float16* fused_h   = (_Float16*)alloc((size_t)B * N * 128 * 2);
  float*    out_feat  = (float*)alloc((size_t)B * N * 64 * 4);
  int*      fps1      = (int*)alloc((size_t)B * 512 * 4);
  float*    new_xyz1  = (float*)alloc((size_t)B * 512 * 3 * 4);
  int*      knn1      = (int*)alloc((size_t)B * 512 * 32 * 4);
  float*    f0_t      = (float*)alloc((size_t)B * 512 * 128 * 4);
  int*      fps2      = (int*)alloc((size_t)B * 256 * 4);
  float*    new_xyz2  = (float*)alloc((size_t)B * 256 * 3 * 4);
  int*      knn2      = (int*)alloc((size_t)B * 256 * 32 * 4);
  _Float16* skip_h    = (_Float16*)alloc((size_t)B * 256 * 64 * 2);
  _Float16* fcat      = (_Float16*)alloc((size_t)B * 256 * 1280 * 2);

  auto cvt = [&](const float* src, _Float16* dst, int n) {
    cvt_f16_kernel<<<(n + 255) / 256, 256, 0, stream>>>(src, dst, n);
  };
  cvt(geo_w, wh_geo, 64 * 128);
  cvt(feat_w, wh_feat, 64 * 128);
  cvt(fuse_w, wh_fuse, 64 * 128);
  cvt(gl0_w, wh_gl0, 128 * 128);
  cvt(gl1_w, wh_gl1, 256 * 256);
  cvt(w2, wh_w2, 1024 * 64);
  cvt(wf, wh_wf, 512 * 1280);

  auto bnp = [&](const float* p, float* s, int C) {
    bn_prep_kernel<<<(C + 255) / 256, 256, 0, stream>>>(p, s, C);
  };
  bnp(bn1, sc_bn1, 64);   bnp(bn2, sc_bn2, 1024);
  bnp(geo_bn, sc_geo, 64); bnp(feat_bn, sc_feat, 64);
  bnp(fuse_bn, sc_fuse, 64); bnp(gl0_bn, sc_gl0, 128);
  bnp(gl1_bn, sc_gl1, 256); bnp(bnf, sc_bnf, 512);

  pointwise1_kernel<<<(B * N + 255) / 256, 256, 0, stream>>>(x, w1, sc_bn1, f_t, B, N);
  extract_xyz_kernel<<<(B * N + 255) / 256, 256, 0, stream>>>(x, xyz_t, B, N);

  // dual graph fusion: geometric + feature KNN (k=20)
  knn_kernel<20, 3><<<(B * N + 255) / 256, 256, 0, stream>>>(xyz_t, xyz_t, N, N, idx_g, B * N);
  knn_kernel<20, 64><<<(B * N + 255) / 256, 256, 0, stream>>>(f_t, f_t, N, N, idx_f, B * N);

  edge_max_gemm<128, 64, 20, true><<<dim3(N / 4, B), 128, 0, stream>>>(
      f_t, N, N, idx_g, nullptr, wh_geo, sc_geo, nullptr, fused_h, 128, 0);
  edge_max_gemm<128, 64, 20, true><<<dim3(N / 4, B), 128, 0, stream>>>(
      f_t, N, N, idx_f, nullptr, wh_feat, sc_feat, nullptr, fused_h, 128, 64);

  gemm_bn_act<128, 64, true><<<dim3(N / 16, B), 32, 0, stream>>>(
      fused_h, N, wh_fuse, sc_fuse, out_feat, nullptr, 0, 0, nullptr);

  // stage 1: FPS 2048 -> 512, knn32, local_op0 (128 -> 128)
  fps_kernel<<<B, 256, 0, stream>>>(xyz_t, N, 512, fps1);
  gather_rows_kernel<<<(B * 512 + 255) / 256, 256, 0, stream>>>(xyz_t, fps1, new_xyz1, N, 512, 3, B * 512);
  knn_kernel<32, 3><<<(B * 512 + 255) / 256, 256, 0, stream>>>(new_xyz1, xyz_t, 512, N, knn1, B * 512);
  edge_max_gemm<128, 128, 32, false><<<dim3(512 / 4, B), 128, 0, stream>>>(
      out_feat, N, 512, knn1, fps1, wh_gl0, sc_gl0, f0_t, nullptr, 0, 0);

  // stage 2: FPS 512 -> 256, knn32, local_op1 (256 -> 256) -> fcat[:, 0:256]
  fps_kernel<<<B, 256, 0, stream>>>(new_xyz1, 512, 256, fps2);
  gather_rows_kernel<<<(B * 256 + 255) / 256, 256, 0, stream>>>(new_xyz1, fps2, new_xyz2, 512, 256, 3, B * 256);
  knn_kernel<32, 3><<<(B * 256 + 255) / 256, 256, 0, stream>>>(new_xyz2, new_xyz1, 256, 512, knn2, B * 256);
  edge_max_gemm<256, 256, 32, false><<<dim3(256 / 4, B), 128, 0, stream>>>(
      f0_t, 512, 256, knn2, fps2, wh_gl1, sc_gl1, nullptr, fcat, 1280, 0);

  // x_skip: strided points -> w2 (64 -> 1024) -> fcat[:, 256:1280]
  skip_gather_kernel<<<(B * 256 * 64 + 255) / 256, 256, 0, stream>>>(f_t, skip_h, B * 256 * 64);
  gemm_bn_act<64, 1024, false><<<dim3(256 / 16, B), 32, 0, stream>>>(
      skip_h, 256, wh_w2, sc_bn2, nullptr, fcat, 1280, 256, nullptr);

  // final: (B,256,1280) x wf^T -> leaky bn -> (B,512,256) channel-major
  gemm_bn_act<1280, 512, true><<<dim3(256 / 16, B), 32, 0, stream>>>(
      fcat, 256, wh_wf, sc_bnf, nullptr, nullptr, 0, 0, out);
}
